// FastESheafLayer_38998303048205
// MI455X (gfx1250) — matrix operand order repaired and checked by reading, not compiled, not run on hardware
//
#include <hip/hip_runtime.h>
#include <hip/hip_bf16.h>

typedef __bf16 bf16_t;
typedef __attribute__((ext_vector_type(8)))  float  v8f;
typedef __attribute__((ext_vector_type(16))) bf16_t v16bf;
typedef __attribute__((ext_vector_type(8)))  bf16_t v8bf;

#define N_NODES 50000
#define N_EDGES 800000
#define DIMX 64
#define DIMY 32
#define NS   64
#define N_MID 6
#define NT   32            // nodes per block
#define MLP_THREADS 128    // 4 wave32 waves

// ---------------------------------------------------------------- WMMA helpers

__device__ inline v16bf make16(v8bf lo, v8bf hi) {
  return __builtin_shufflevector(lo, hi, 0,1,2,3,4,5,6,7,8,9,10,11,12,13,14,15);
}

// A matrix 16x32 bf16 from row-major [rows][64] buffer.
// Lane layout (ISA 7.12.2): m = row0 + (lane&15); half = lane>>4;
// element e -> k = (e&7) + 8*half + 16*(e>>3)  (within the 32-wide k slab)
__device__ inline v16bf load_tileA(const bf16_t* base, int row0, int kslab, int lane) {
  int lh = lane >> 4;
  int m  = row0 + (lane & 15);
  const bf16_t* p = base + m * 64 + kslab * 32 + lh * 8;
  v8bf lo = *(const v8bf*)p;          // k = 8*half + 0..7
  v8bf hi = *(const v8bf*)(p + 16);   // k = 16 + 8*half + 0..7
  return make16(lo, hi);
}

// B matrix 32x16 bf16 from transposed weights wT[col][64] (contiguous in k).
// Lane layout: n = lane&15; element e -> k = (e&7) + 8*half + 16*(e>>3)
__device__ inline v16bf load_tileB(const bf16_t* base, int c0, int kslab, int lane) {
  int lh = lane >> 4;
  int c  = c0 + (lane & 15);
  const bf16_t* p = base + c * 64 + kslab * 32 + lh * 8;
  v8bf lo = *(const v8bf*)p;          // k = 8*half + 0..7
  v8bf hi = *(const v8bf*)(p + 16);   // k = 16 + 8*half + 0..7
  return make16(lo, hi);
}

__device__ inline v8f wmma_bf16(v16bf a, v16bf b, v8f c) {
  return __builtin_amdgcn_wmma_f32_16x16x32_bf16(false, a, false, b, (short)0, c, false, false);
}

// ---------------------------------------------------------------- prep kernels

__global__ void prep_small_kernel(const float* __restrict__ w_in,
                                  const float* __restrict__ w_mid,
                                  bf16_t* __restrict__ w_inT,
                                  bf16_t* __restrict__ w_midT) {
  int t = blockIdx.x * blockDim.x + threadIdx.x;
  if (t >= 7 * NS * NS) return;
  int mat = t >> 12;          // 0 = w_in, 1..6 = w_mid
  int rem = t & 4095;
  int k = rem >> 6, c = rem & 63;
  float v = (mat == 0) ? w_in[k * NS + c] : w_mid[(mat - 1) * NS * NS + k * NS + c];
  bf16_t* dst = (mat == 0) ? w_inT : (w_midT + (mat - 1) * NS * NS);
  dst[c * NS + k] = (bf16_t)v;
}

__global__ void prep_out_kernel(const float* __restrict__ w_out,
                                bf16_t* __restrict__ w_outT) {
  int t = blockIdx.x * blockDim.x + threadIdx.x;
  if (t >= NS * DIMY * DIMX) return;   // 131072
  int k = t >> 11;                     // / 2048
  int c = t & 2047;
  w_outT[c * NS + k] = (bf16_t)w_out[t];
}

__global__ void zero_kernel(float* __restrict__ out, float* __restrict__ deg) {
  int t = blockIdx.x * blockDim.x + threadIdx.x;
  if (t < N_NODES * DIMY) out[t] = 0.f;
  if (t < N_NODES)        deg[t] = 0.f;
}

__global__ void deg_kernel(const long long* __restrict__ ei, float* __restrict__ deg) {
  int e = blockIdx.x * blockDim.x + threadIdx.x;
  if (e >= N_EDGES) return;
  int c = (int)ei[N_EDGES + e];
  atomicAdd(&deg[c], 1.0f);
}

__global__ void dinv_kernel(const float* __restrict__ deg, float* __restrict__ dinv) {
  int n = blockIdx.x * blockDim.x + threadIdx.x;
  if (n >= N_NODES) return;
  float d = deg[n];
  dinv[n] = d > 0.f ? rsqrtf(fmaxf(d, 1.f)) : 0.f;
}

// ---------------------------------------------------------------- main kernel

__global__ __launch_bounds__(MLP_THREADS)
void sheaf_mlp_kernel(const float* __restrict__ xf,
                      const float* __restrict__ b_in,
                      const float* __restrict__ b_mid,
                      const float* __restrict__ b_out,
                      const bf16_t* __restrict__ w_inT,
                      const bf16_t* __restrict__ w_midT,
                      const bf16_t* __restrict__ w_outT,
                      float* __restrict__ q) {
  __shared__ __align__(16) float  xfF[NT * 64];
  __shared__ __align__(16) bf16_t xfB[NT * 64];
  __shared__ __align__(16) bf16_t hbuf[2][NT * 64];

  const int nbase = blockIdx.x * NT;
  const int lane  = threadIdx.x & 31;
  const int wave  = threadIdx.x >> 5;
  const int lh    = lane >> 4;
  const int l15   = lane & 15;

  // stage node-feature tile (f32 for the bilinear reduce, bf16 for layer-0 A)
  for (int i = threadIdx.x; i < NT * 64; i += MLP_THREADS) {
    int n = nbase + (i >> 6);
    if (n >= N_NODES) n = N_NODES - 1;
    float v = xf[n * 64 + (i & 63)];
    xfF[i] = v;
    xfB[i] = (bf16_t)v;
  }
  __syncthreads();

  const int c0 = wave * 16;  // this wave's column tile in the 64-wide MLP

  // ---- layer 0: relu(x @ w_in + b_in) -> hbuf[0]
  {
    v16bf B0 = load_tileB(w_inT, c0, 0, lane);
    v16bf B1 = load_tileB(w_inT, c0, 1, lane);
    float bias = b_in[c0 + l15];
    #pragma unroll
    for (int rt = 0; rt < NT / 16; ++rt) {
      v16bf A0 = load_tileA(xfB, rt * 16, 0, lane);
      v16bf A1 = load_tileA(xfB, rt * 16, 1, lane);
      v8f acc = {};
      acc = wmma_bf16(A0, B0, acc);
      acc = wmma_bf16(A1, B1, acc);
      #pragma unroll
      for (int e = 0; e < 8; ++e) {
        float f = acc[e] + bias;
        f = f > 0.f ? f : 0.f;
        hbuf[0][(rt * 16 + e + 8 * lh) * 64 + c0 + l15] = (bf16_t)f;
      }
    }
  }
  __syncthreads();

  // ---- six mid layers, ping-pong through LDS
  for (int L = 0; L < N_MID; ++L) {
    const bf16_t* hin  = hbuf[L & 1];
    bf16_t*       hout = hbuf[(L + 1) & 1];
    const bf16_t* wT   = w_midT + L * NS * NS;
    v16bf B0 = load_tileB(wT, c0, 0, lane);
    v16bf B1 = load_tileB(wT, c0, 1, lane);
    float bias = b_mid[L * NS + c0 + l15];
    #pragma unroll
    for (int rt = 0; rt < NT / 16; ++rt) {
      v16bf A0 = load_tileA(hin, rt * 16, 0, lane);
      v16bf A1 = load_tileA(hin, rt * 16, 1, lane);
      v8f acc = {};
      acc = wmma_bf16(A0, B0, acc);
      acc = wmma_bf16(A1, B1, acc);
      #pragma unroll
      for (int e = 0; e < 8; ++e) {
        float f = acc[e] + bias;
        f = f > 0.f ? f : 0.f;
        hout[(rt * 16 + e + 8 * lh) * 64 + c0 + l15] = (bf16_t)f;
      }
    }
    __syncthreads();
  }

  // ---- fused bilinear: q[n][y] = sum_x (h.w_out + b_out)[n][y*64+x] * xf[n][x]
  const bf16_t* hfin = hbuf[N_MID & 1];
  v16bf HA[NT / 16][2];
  #pragma unroll
  for (int rt = 0; rt < NT / 16; ++rt) {
    HA[rt][0] = load_tileA(hfin, rt * 16, 0, lane);
    HA[rt][1] = load_tileA(hfin, rt * 16, 1, lane);
  }
  // node features pre-staged in the WMMA C layout: element (vgpr e, lane) = xf[n][x]
  v8f XC[NT / 16][4];
  #pragma unroll
  for (int rt = 0; rt < NT / 16; ++rt)
    #pragma unroll
    for (int xt = 0; xt < 4; ++xt)
      #pragma unroll
      for (int e = 0; e < 8; ++e)
        XC[rt][xt][e] = xfF[(rt * 16 + e + 8 * lh) * 64 + xt * 16 + l15];

  for (int yy = 0; yy < 8; ++yy) {
    int yg = wave * 8 + yy;               // this wave's output row y
    v8f qa[NT / 16];
    #pragma unroll
    for (int rt = 0; rt < NT / 16; ++rt) qa[rt] = (v8f){};
    #pragma unroll
    for (int xt = 0; xt < 4; ++xt) {
      int cc = yg * 64 + xt * 16;
      v16bf B0 = load_tileB(w_outT, cc, 0, lane);
      v16bf B1 = load_tileB(w_outT, cc, 1, lane);
      float bo = b_out[cc + l15];
      #pragma unroll
      for (int rt = 0; rt < NT / 16; ++rt) {
        v8f acc;
        #pragma unroll
        for (int e = 0; e < 8; ++e) acc[e] = bo;   // fold b_out into the C matrix
        acc = wmma_bf16(HA[rt][0], B0, acc);
        acc = wmma_bf16(HA[rt][1], B1, acc);
        #pragma unroll
        for (int e = 0; e < 8; ++e) qa[rt][e] += acc[e] * XC[rt][xt][e];
      }
    }
    // reduce the 16 tile columns (lanes 0-15 / 16-31 independently)
    #pragma unroll
    for (int rt = 0; rt < NT / 16; ++rt) {
      #pragma unroll
      for (int e = 0; e < 8; ++e) {
        float r = qa[rt][e];
        r += __shfl_xor(r, 1, 32);
        r += __shfl_xor(r, 2, 32);
        r += __shfl_xor(r, 4, 32);
        r += __shfl_xor(r, 8, 32);
        int n = nbase + rt * 16 + e + 8 * lh;
        if (l15 == 0 && n < N_NODES) q[n * DIMY + yg] = r;
      }
    }
  }
}

// ---------------------------------------------------------------- edge scatter

__global__ void scatter_kernel(const long long* __restrict__ ei,
                               const float* __restrict__ dinv,
                               const float* __restrict__ q,
                               float* __restrict__ out) {
  int t = blockIdx.x * blockDim.x + threadIdx.x;
  if (t >= N_EDGES * DIMY) return;
  int e = t >> 5;          // one wave per edge, lane = y component
  int y = t & 31;
  int r = (int)ei[e];
  int c = (int)ei[N_EDGES + e];
  float nm = dinv[r] * dinv[c];
  atomicAdd(&out[c * DIMY + y], nm * q[r * DIMY + y]);
}

// ---------------------------------------------------------------- launch

extern "C" void kernel_launch(void* const* d_in, const int* in_sizes, int n_in,
                              void* d_out, int out_size, void* d_ws, size_t ws_size,
                              hipStream_t stream) {
  const float*     xf    = (const float*)d_in[0];
  const long long* ei    = (const long long*)d_in[1];   // int64 edge_index [2][E]
  const float*     w_in  = (const float*)d_in[2];
  const float*     b_in  = (const float*)d_in[3];
  const float*     w_mid = (const float*)d_in[4];
  const float*     b_mid = (const float*)d_in[5];
  const float*     w_out = (const float*)d_in[6];
  const float*     b_out = (const float*)d_in[7];
  float* out = (float*)d_out;

  // workspace layout (all 16B aligned)
  float*  q      = (float*)d_ws;                    // 50000*32 f32
  float*  deg    = q + N_NODES * DIMY;              // 50000 f32
  float*  dinv   = deg + N_NODES;                   // 50000 f32
  bf16_t* w_inT  = (bf16_t*)(dinv + N_NODES);       // 64*64 bf16 (transposed)
  bf16_t* w_midT = w_inT + NS * NS;                 // 6*64*64 bf16
  bf16_t* w_outT = w_midT + N_MID * NS * NS;        // 2048*64 bf16

  prep_small_kernel<<<(7 * NS * NS + 255) / 256, 256, 0, stream>>>(w_in, w_mid, w_inT, w_midT);
  prep_out_kernel<<<(NS * DIMY * DIMX + 255) / 256, 256, 0, stream>>>(w_out, w_outT);
  zero_kernel<<<(N_NODES * DIMY + 255) / 256, 256, 0, stream>>>(out, deg);
  deg_kernel<<<(N_EDGES + 255) / 256, 256, 0, stream>>>(ei, deg);
  dinv_kernel<<<(N_NODES + 255) / 256, 256, 0, stream>>>(deg, dinv);

  int nblk = (N_NODES + NT - 1) / NT;   // 1563 blocks x 4 waves
  sheaf_mlp_kernel<<<nblk, MLP_THREADS, 0, stream>>>(xf, b_in, b_mid, b_out,
                                                     w_inT, w_midT, w_outT, q);

  scatter_kernel<<<(N_EDGES * DIMY + 255) / 256, 256, 0, stream>>>(ei, dinv, q, out);
}
